// GroupedDilatedAttention_GlobalQKV_18588618457407
// MI455X (gfx1250) — compile-verified, matrix operand order checked by reading, not run on hardware
//
#include <hip/hip_runtime.h>
#include <hip/hip_bf16.h>

typedef __bf16 v16bf __attribute__((ext_vector_type(16)));
typedef float  v8f   __attribute__((ext_vector_type(8)));

// ---------- bf16 helpers ----------
__device__ __forceinline__ unsigned short f2bfu(float f) {
    union { float f; unsigned u; } a; a.f = f;
    return (unsigned short)((a.u + 0x8000u) >> 16);
}
// pack two f32 -> dword of 2 bf16 (2x v_add + v_perm_b32)
__device__ __forceinline__ unsigned pack2bf(float lo, float hi) {
    union { float f; unsigned u; } a, b; a.f = lo; b.f = hi;
    return __builtin_amdgcn_perm(b.u + 0x8000u, a.u + 0x8000u, 0x07060302u);
}
__device__ __forceinline__ float bflo(unsigned u) {
    union { unsigned u; float f; } a; a.u = u << 16; return a.f;
}
__device__ __forceinline__ float bfhi(unsigned u) {
    union { unsigned u; float f; } a; a.u = u & 0xFFFF0000u; return a.f;
}

// Build 16x32 bf16 fragment from a global fp32 row (convert on the fly).
// ISA 16-bit layout: lane half=0 holds K {0..7,16..23}, half=1 holds {8..15,24..31}.
__device__ __forceinline__ v16bf load_frag_f32row(const float* __restrict__ rowk,
                                                  unsigned half, bool valid) {
    union { v16bf v; unsigned u[8]; } r;
    if (valid) {
        const float4* p0 = reinterpret_cast<const float4*>(rowk + half * 8);
        const float4* p1 = reinterpret_cast<const float4*>(rowk + 16 + half * 8);
        float4 a0 = p0[0], a1 = p0[1], b0 = p1[0], b1 = p1[1];
        r.u[0] = pack2bf(a0.x, a0.y); r.u[1] = pack2bf(a0.z, a0.w);
        r.u[2] = pack2bf(a1.x, a1.y); r.u[3] = pack2bf(a1.z, a1.w);
        r.u[4] = pack2bf(b0.x, b0.y); r.u[5] = pack2bf(b0.z, b0.w);
        r.u[6] = pack2bf(b1.x, b1.y); r.u[7] = pack2bf(b1.z, b1.w);
    } else {
#pragma unroll
        for (int e = 0; e < 8; ++e) r.u[e] = 0u;
    }
    return r.v;
}

// Fragment from pre-converted bf16 row in GLOBAL memory: two b128 loads, no VALU.
__device__ __forceinline__ v16bf load_frag_bf16row(const unsigned short* __restrict__ rowk,
                                                   unsigned half) {
    union { v16bf v; uint4 q[2]; } r;
    r.q[0] = *reinterpret_cast<const uint4*>(rowk + half * 8);
    r.q[1] = *reinterpret_cast<const uint4*>(rowk + 16 + half * 8);
    return r.v;
}
// Same but from LDS (two ds_load_b128).
__device__ __forceinline__ v16bf load_frag_lds(const unsigned short* __restrict__ basek,
                                               unsigned half) {
    union { v16bf v; uint4 q[2]; } r;
    r.q[0] = *reinterpret_cast<const uint4*>(basek + half * 8);
    r.q[1] = *reinterpret_cast<const uint4*>(basek + 16 + half * 8);
    return r.v;
}

// ------------------- constants -------------------
#define TH 8
#define TW 16
#define HALO 3
#define PH (TH + 2*HALO)        // 14
#define PW (TW + 2*HALO)        // 22
#define NPIX (PH * PW)          // 308
#define MT_KV ((NPIX + 15)/16)  // 20
#define KSTR 132                // LDS row stride (shorts)
#define KV_SZ (NPIX * KSTR)
#define Q_SZ  (128 * KSTR)
#define LDS1_SHORTS (2*KV_SZ + Q_SZ)   // 196416 B

#define WQ_ELEMS (1152 * 384)
#define WP_ELEMS (384 * 384)

// ---------------- Kernel 0: fp32 -> packed bf16 weight conversion ----------------
__global__ void __launch_bounds__(256)
k_cvt(const float* __restrict__ src, unsigned* __restrict__ dst, int npairs) {
    const int i = blockIdx.x * 256 + threadIdx.x;
    if (i < npairs) {
        const float2 v = reinterpret_cast<const float2*>(src)[i];
        dst[i] = pack2bf(v.x, v.y);
    }
}

// =====================================================================
// Kernel 1: fused QKV projection (bf16 WMMA) + dilated 9-tap attention.
// PREW: weights pre-converted to bf16 in global scratch.
// grid = (128 tiles, 3 dilations, 4 batches), block = 256 (8 wave32)
// =====================================================================
template<bool PREW>
__global__ void __launch_bounds__(256)
k_qkv_attn(const float* __restrict__ x, const float* __restrict__ wqkv,
           const unsigned short* __restrict__ wq16, float* __restrict__ out) {
    extern __shared__ unsigned short sm[];
    unsigned short* kB = sm;
    unsigned short* vB = sm + KV_SZ;
    unsigned short* qB = sm + 2 * KV_SZ;

    const int bz   = blockIdx.z;
    const int di   = blockIdx.y;
    const int tile = blockIdx.x;
    const int tx = tile & 7, ty = tile >> 3;
    const int h0 = ty * TH, w0 = tx * TW;

    const int tid  = threadIdx.x;
    const unsigned lane = tid & 31;
    const unsigned wave = tid >> 5;
    const unsigned m    = lane & 15;
    const unsigned half = lane >> 4;

    const float* xb = x + (size_t)bz * (128u * 128u * 384u);

    // ---------- Phase 1: K/V GEMM over padded tile (20 Mtiles x 16 Ntiles) ----------
    for (int mt = wave; mt < MT_KV; mt += 8) {
        const int pixel = mt * 16 + (int)m;
        const int ph = pixel / PW, pw = pixel - ph * PW;
        const int gh = h0 - HALO + ph, gw = w0 - HALO + pw;
        const bool valid = (pixel < NPIX) && gh >= 0 && gh < 128 && gw >= 0 && gw < 128;
        const float* xrow = xb + ((size_t)(gh * 128 + gw)) * 384u;

        v16bf afr[12];
#pragma unroll
        for (int ks = 0; ks < 12; ++ks)
            afr[ks] = load_frag_f32row(xrow + ks * 32, half, valid);

        for (int nt = 0; nt < 16; ++nt) {
            const size_t wro = (size_t)(di * 384 + 128 + nt * 16 + (int)m) * 384u;
            v8f acc = {0.f,0.f,0.f,0.f,0.f,0.f,0.f,0.f};
#pragma unroll
            for (int ks = 0; ks < 12; ++ks) {
                v16bf bf = PREW ? load_frag_bf16row(wq16 + wro + ks * 32, half)
                                : load_frag_f32row(wqkv + wro + ks * 32, half, true);
                acc = __builtin_amdgcn_wmma_f32_16x16x32_bf16(
                    false, afr[ks], false, bf, (short)0, acc, false, false);
            }
            const int chan = nt * 16 + (int)m;
            unsigned short* dst = (chan < 128) ? (kB + chan) : (vB + (chan - 128));
#pragma unroll
            for (int r = 0; r < 8; ++r) {
                const int px = mt * 16 + r + 8 * (int)half;
                if (px < NPIX) dst[(size_t)px * KSTR] = f2bfu(acc[r]);
            }
        }
    }

    // ---------- Phase 2: Q GEMM over inner tile (8 Mtiles x 8 Ntiles) ----------
    {
        const int mt = wave;
        const int pi = mt * 16 + (int)m;
        const int iph = pi >> 4, ipw = pi & 15;
        const float* xrow = xb + ((size_t)((h0 + iph) * 128 + (w0 + ipw))) * 384u;

        v16bf afr[12];
#pragma unroll
        for (int ks = 0; ks < 12; ++ks)
            afr[ks] = load_frag_f32row(xrow + ks * 32, half, true);

        for (int nt = 0; nt < 8; ++nt) {
            const size_t wro = (size_t)(di * 384 + nt * 16 + (int)m) * 384u;
            v8f acc = {0.f,0.f,0.f,0.f,0.f,0.f,0.f,0.f};
#pragma unroll
            for (int ks = 0; ks < 12; ++ks) {
                v16bf bf = PREW ? load_frag_bf16row(wq16 + wro + ks * 32, half)
                                : load_frag_f32row(wqkv + wro + ks * 32, half, true);
                acc = __builtin_amdgcn_wmma_f32_16x16x32_bf16(
                    false, afr[ks], false, bf, (short)0, acc, false, false);
            }
#pragma unroll
            for (int r = 0; r < 8; ++r) {
                const int px = mt * 16 + r + 8 * (int)half;
                qB[(size_t)px * KSTR + nt * 16 + (int)m] = f2bfu(acc[r]);
            }
        }
    }

    __syncthreads();

    // ---------- Phase 3: 9-tap dilated attention ----------
    const float scale = 0.17677669529663688f;   // 32^-0.5
    const int rd = di + 1;
#pragma unroll
    for (int s = 0; s < 2; ++s) {
        const int task = tid + 256 * s;           // 128 pixels * 4 heads
        const int p = task >> 2, j = task & 3;
        const int iph = p >> 4, ipw = p & 15;

        const unsigned* q32 = reinterpret_cast<const unsigned*>(qB + (size_t)p * KSTR + j * 32);
        float qv[32];
#pragma unroll
        for (int d = 0; d < 16; ++d) {
            const unsigned u = q32[d];
            qv[2*d] = bflo(u); qv[2*d+1] = bfhi(u);
        }

        const int cph = iph + HALO, cpw = ipw + HALO;
        float lg[9]; int pp[9];
#pragma unroll
        for (int t = 0; t < 9; ++t) {
            const int dh = (t / 3 - 1) * rd, dw = (t % 3 - 1) * rd;
            const int q_ = (cph + dh) * PW + (cpw + dw);
            pp[t] = q_;
            const unsigned* kr = reinterpret_cast<const unsigned*>(kB + (size_t)q_ * KSTR + j * 32);
            float d = 0.f;
#pragma unroll
            for (int dd = 0; dd < 16; ++dd) {
                const unsigned u = kr[dd];
                d += qv[2*dd] * bflo(u) + qv[2*dd+1] * bfhi(u);
            }
            lg[t] = d * scale;
        }
        float mx = lg[0];
#pragma unroll
        for (int t = 1; t < 9; ++t) mx = fmaxf(mx, lg[t]);
        float wts[9]; float sum = 0.f;
#pragma unroll
        for (int t = 0; t < 9; ++t) { wts[t] = __expf(lg[t] - mx); sum += wts[t]; }
        const float inv = 1.f / sum;
#pragma unroll
        for (int t = 0; t < 9; ++t) wts[t] *= inv;

        float2* dst = reinterpret_cast<float2*>(
            out + (size_t)(bz * 12 + di * 4 + j) * 524288u
                + (size_t)((h0 + iph) * 128 + (w0 + ipw)) * 32u);
#pragma unroll
        for (int dd = 0; dd < 16; ++dd) {
            float alo = 0.f, ahi = 0.f;
#pragma unroll
            for (int t = 0; t < 9; ++t) {
                const unsigned u = *reinterpret_cast<const unsigned*>(
                    vB + (size_t)pp[t] * KSTR + j * 32 + 2 * dd);
                alo += wts[t] * bflo(u);
                ahi += wts[t] * bfhi(u);
            }
            dst[dd] = make_float2(alo, ahi);
        }
    }
}

// =====================================================================
// Kernel 2: in-place output projection  out[r,:] = out[r,:] @ Wp^T + b
// grid = 512 blocks x 128 rows, block = 256 (8 wave32)
// =====================================================================
#define PSTR 392   // LDS row stride in shorts (16B-aligned rows)

template<bool PREW>
__global__ void __launch_bounds__(256)
k_proj(float* __restrict__ io, const float* __restrict__ wproj,
       const unsigned short* __restrict__ wp16, const float* __restrict__ bias) {
    extern __shared__ unsigned short sm2[];
    const int rowBase = blockIdx.x * 128;
    const int tid  = threadIdx.x;
    const unsigned lane = tid & 31;
    const unsigned wave = tid >> 5;
    const unsigned m    = lane & 15;
    const unsigned half = lane >> 4;

    // Stage 128 rows x 384 ch into LDS as packed bf16 pairs (coalesced float2).
    {
        const float2* io2 = reinterpret_cast<const float2*>(io);
        unsigned* smd = reinterpret_cast<unsigned*>(sm2);
        for (int e = tid; e < 128 * 192; e += 256) {
            const int rr = e / 192, c2 = e - rr * 192;
            const float2 v = io2[(size_t)(rowBase + rr) * 192u + c2];
            smd[rr * (PSTR/2) + c2] = pack2bf(v.x, v.y);
        }
    }
    __syncthreads();

    {
        const int mt = wave;
        const unsigned short* arow = sm2 + (size_t)(mt * 16 + (int)m) * PSTR;
        v16bf afr[12];
#pragma unroll
        for (int ks = 0; ks < 12; ++ks)
            afr[ks] = load_frag_lds(arow + ks * 32, half);

        for (int nt = 0; nt < 24; ++nt) {
            const size_t wro = (size_t)(nt * 16 + (int)m) * 384u;
            v8f acc = {0.f,0.f,0.f,0.f,0.f,0.f,0.f,0.f};
#pragma unroll
            for (int ks = 0; ks < 12; ++ks) {
                v16bf bf = PREW ? load_frag_bf16row(wp16 + wro + ks * 32, half)
                                : load_frag_f32row(wproj + wro + ks * 32, half, true);
                acc = __builtin_amdgcn_wmma_f32_16x16x32_bf16(
                    false, afr[ks], false, bf, (short)0, acc, false, false);
            }
            const float bv = bias[nt * 16 + (int)m];
#pragma unroll
            for (int r = 0; r < 8; ++r) {
                const int row = rowBase + mt * 16 + r + 8 * (int)half;
                io[(size_t)row * 384u + nt * 16 + (int)m] = acc[r] + bv;
            }
        }
    }
}

extern "C" void kernel_launch(void* const* d_in, const int* in_sizes, int n_in,
                              void* d_out, int out_size, void* d_ws, size_t ws_size,
                              hipStream_t stream) {
    (void)in_sizes; (void)n_in; (void)out_size;
    const float* x     = (const float*)d_in[0];
    const float* wqkv  = (const float*)d_in[1];
    const float* wproj = (const float*)d_in[2];
    const float* bproj = (const float*)d_in[3];
    float* out = (float*)d_out;

    unsigned short* wq16 = (unsigned short*)d_ws;
    unsigned short* wp16 = wq16 + WQ_ELEMS;
    const bool prew = (d_ws != nullptr) &&
                      (ws_size >= (size_t)(WQ_ELEMS + WP_ELEMS) * sizeof(unsigned short));

    dim3 g1(128, 3, 4);
    const size_t lds1 = LDS1_SHORTS * sizeof(unsigned short);
    const size_t lds2 = 128 * PSTR * sizeof(unsigned short);

    if (prew) {
        k_cvt<<<(WQ_ELEMS/2 + 255) / 256, 256, 0, stream>>>(wqkv,  (unsigned*)wq16, WQ_ELEMS/2);
        k_cvt<<<(WP_ELEMS/2 + 255) / 256, 256, 0, stream>>>(wproj, (unsigned*)wp16, WP_ELEMS/2);
        k_qkv_attn<true><<<g1, 256, lds1, stream>>>(x, wqkv, wq16, out);
        k_proj<true><<<512, 256, lds2, stream>>>(out, wproj, wp16, bproj);
    } else {
        k_qkv_attn<false><<<g1, 256, lds1, stream>>>(x, wqkv, wq16, out);
        k_proj<false><<<512, 256, lds2, stream>>>(out, wproj, wp16, bproj);
    }
}
